// SPConvModel_32298154066080
// MI455X (gfx1250) — matrix operand order repaired and checked by reading, not compiled, not run on hardware
//
#include <hip/hip_runtime.h>

#define NPTS 250000
#define NSUP 10000
#define NEDG 320000
#define DD   128
#define DEE  64
#define LEPS 1e-5f

typedef __attribute__((ext_vector_type(16))) __bf16 v16bf;
typedef __attribute__((ext_vector_type(8)))  float  v8f;
typedef __attribute__((ext_vector_type(4)))  unsigned int v4u;

union FragAB { v16bf bf; v4u u4[2]; };

static __device__ __forceinline__ unsigned short f2bf(float f) {
  unsigned int u = __float_as_uint(f);
  u += 0x7FFFu + ((u >> 16) & 1u);
  return (unsigned short)(u >> 16);
}

// fused (sum, sumsq) reduction over 128 threads: one barrier tree instead of two
static __device__ __forceinline__ float2 block_sum2_128(float a, float b,
                                                        float* redA, float* redB, int t) {
  redA[t] = a; redB[t] = b; __syncthreads();
  #pragma unroll
  for (int o = 64; o > 0; o >>= 1) {
    if (t < o) { redA[t] += redA[t + o]; redB[t] += redB[t + o]; }
    __syncthreads();
  }
  float2 r = make_float2(redA[0], redB[0]); __syncthreads();
  return r;
}

// ---------------- centroids / degrees ----------------
__global__ void cent_acc_k(const float* __restrict__ pos, const int* __restrict__ sidx,
                           float* __restrict__ csum) {
  int i = blockIdx.x * 256 + threadIdx.x;
  if (i >= NPTS) return;
  int s = sidx[i];
  atomicAdd(&csum[s * 4 + 0], pos[i * 3 + 0]);
  atomicAdd(&csum[s * 4 + 1], pos[i * 3 + 1]);
  atomicAdd(&csum[s * 4 + 2], pos[i * 3 + 2]);
  atomicAdd(&csum[s * 4 + 3], 1.0f);
}

__global__ void cent_fin_k(const float* __restrict__ csum, float* __restrict__ cent) {
  int s = blockIdx.x * 256 + threadIdx.x;
  if (s >= NSUP) return;
  float c = fmaxf(csum[s * 4 + 3], 1.0f);
  cent[s * 3 + 0] = csum[s * 4 + 0] / c;
  cent[s * 3 + 1] = csum[s * 4 + 1] / c;
  cent[s * 3 + 2] = csum[s * 4 + 2] / c;
}

__global__ void deg_acc_k(const int* __restrict__ dst, float* __restrict__ cnt) {
  int i = blockIdx.x * 256 + threadIdx.x;
  if (i >= NEDG) return;
  atomicAdd(&cnt[dst[i]], 1.0f);
}

// ---------------- PointNet encoder + segment-max ----------------
__global__ void encoder_k(const float* __restrict__ pos, const float* __restrict__ feats,
                          const int* __restrict__ sidx, const float* __restrict__ cent,
                          const float* __restrict__ w1, const float* __restrict__ b1,
                          const float* __restrict__ g1, const float* __restrict__ be1,
                          const float* __restrict__ w2, const float* __restrict__ b2,
                          const float* __restrict__ g2, const float* __restrict__ be2,
                          float* __restrict__ x) {
  __shared__ float in9[9];
  __shared__ float h1s[DD];
  __shared__ float redA[DD], redB[DD];
  int p = blockIdx.x, t = threadIdx.x;
  int sp = sidx[p];
  if (t < 9) in9[t] = (t < 3) ? (pos[p * 3 + t] - cent[sp * 3 + t]) : feats[p * 6 + (t - 3)];
  __syncthreads();
  float a = b1[t];
  #pragma unroll
  for (int k = 0; k < 9; ++k) a += in9[k] * w1[k * DD + t];
  float2 r = block_sum2_128(a, a * a, redA, redB, t);
  float m = r.x * (1.0f / 128.0f);
  float v = fmaxf(r.y * (1.0f / 128.0f) - m * m, 0.0f);
  float h = fmaxf((a - m) * rsqrtf(v + LEPS) * g1[t] + be1[t], 0.0f);
  h1s[t] = h;
  __syncthreads();
  float a2 = b2[t];
  for (int k = 0; k < DD; ++k) a2 += h1s[k] * w2[k * DD + t];
  r = block_sum2_128(a2, a2 * a2, redA, redB, t);
  m = r.x * (1.0f / 128.0f);
  v = fmaxf(r.y * (1.0f / 128.0f) - m * m, 0.0f);
  float h2 = fmaxf((a2 - m) * rsqrtf(v + LEPS) * g2[t] + be2[t], 0.0f);
  // post-relu values >= 0, so unsigned-bit compare == float compare; empty segs stay 0
  atomicMax((unsigned int*)&x[(size_t)sp * DD + t], __float_as_uint(h2));
}

// ---------------- fp32 -> bf16 ----------------
__global__ void tobf_k(const float* __restrict__ x, unsigned short* __restrict__ xb, int n) {
  int i = blockIdx.x * 256 + threadIdx.x;
  if (i < n) xb[i] = f2bf(x[i]);
}

// ---------------- pack weights [K][128] f32 -> WMMA-B-striped bf16 ----------------
// out[((kc*8+wv)*32 + lane)*16 + e] = W[(kc*32 + (e&7) + 8*(lane>>4) + 16*(e>>3)) * 128 + wv*16 + (lane&15)]
__global__ void packw_k(const float* __restrict__ W, unsigned short* __restrict__ out, int K) {
  int idx = blockIdx.x * 256 + threadIdx.x;
  if (idx >= K * DD) return;
  int e    = idx & 15;
  int lane = (idx >> 4) & 31;
  int tt   = idx >> 9;          // kc*8 + wv
  int kc   = tt >> 3;
  int wv   = tt & 7;
  int k = kc * 32 + (e & 7) + 8 * (lane >> 4) + 16 * (e >> 3);
  int n = wv * 16 + (lane & 15);
  out[idx] = f2bf(W[k * DD + n]);
}

// ---------------- edge positional MLP: e = relu(relu(delta@W1+b1)@W2+b2) ----------------
__global__ void edgefeat_k(const int* __restrict__ src, const int* __restrict__ dst,
                           const float* __restrict__ cent,
                           const float* __restrict__ w1, const float* __restrict__ b1,
                           const float* __restrict__ w2, const float* __restrict__ b2,
                           unsigned short* __restrict__ ebf) {
  __shared__ float t1[4][DEE];
  int g = threadIdx.x >> 6;
  int c = threadIdx.x & 63;
  int e = blockIdx.x * 4 + g;
  if (e < NEDG) {
    int s = src[e], d = dst[e];
    float dx = cent[d * 3 + 0] - cent[s * 3 + 0];
    float dy = cent[d * 3 + 1] - cent[s * 3 + 1];
    float dz = cent[d * 3 + 2] - cent[s * 3 + 2];
    t1[g][c] = fmaxf(dx * w1[c] + dy * w1[DEE + c] + dz * w1[2 * DEE + c] + b1[c], 0.0f);
  }
  __syncthreads();
  if (e < NEDG) {
    float acc = b2[c];
    #pragma unroll 8
    for (int k = 0; k < DEE; ++k) acc += t1[g][k] * w2[k * DEE + c];
    ebf[(size_t)e * DEE + c] = f2bf(fmaxf(acc, 0.0f));
  }
}

// ---------------- fused edge-message double GEMM (WMMA) + segment-sum ----------------
// block = 256 threads = 8 waves; 32 edges per block (two 16-row tiles per wave sharing
// each B fragment -> 2 WMMAs per B fetch, 2 independent accumulator chains per wave)
__global__ void __launch_bounds__(256)
edgemsg_k(const int* __restrict__ src, const int* __restrict__ dst,
          const unsigned short* __restrict__ xbf, const unsigned short* __restrict__ ebf,
          const unsigned short* __restrict__ W1p, const float* __restrict__ b1,
          const unsigned short* __restrict__ W2p, const float* __restrict__ b2,
          float* __restrict__ Msum) {
  __shared__ unsigned short meA[32 * 320];  // 32 edges x (x_src|x_dst|e)   20 KB
  __shared__ unsigned short hB[32 * DD];    // relu(GEMM1) intermediate      8 KB
  __shared__ int sIdx[32], dIdx[32];

  int tid = threadIdx.x;
  int e0  = blockIdx.x * 32;
  if (tid < 32) {
    int e = min(e0 + tid, NEDG - 1);
    sIdx[tid] = src[e];
    dIdx[tid] = dst[e];
  }
  __syncthreads();

  // stage me rows as dwords (2 bf16 each): 32 rows x 160 dwords
  for (int i = tid; i < 32 * 160; i += 256) {
    int r = i / 160, d = i - r * 160;
    unsigned int val;
    if (d < 64)
      val = *(const unsigned int*)(xbf + (size_t)sIdx[r] * DD + (d << 1));
    else if (d < 128)
      val = *(const unsigned int*)(xbf + (size_t)dIdx[r] * DD + ((d - 64) << 1));
    else
      val = *(const unsigned int*)(ebf + (size_t)min(e0 + r, NEDG - 1) * DEE + ((d - 128) << 1));
    *(unsigned int*)(meA + r * 320 + (d << 1)) = val;
  }
  __syncthreads();

  int wv   = tid >> 5;       // col tile
  int lane = tid & 31;
  int hf   = lane >> 4;      // lane half
  int ln   = lane & 15;
  int col  = wv * 16 + ln;

  // GEMM1: h = relu(me @ W1 + b1), K = 320
  float bv = b1[col];
  v8f acc0 = {bv, bv, bv, bv, bv, bv, bv, bv};
  v8f acc1 = acc0;
  #pragma unroll
  for (int kc = 0; kc < 10; ++kc) {
    FragAB a0, a1, b;
    const v4u* bp = (const v4u*)(W1p + (((size_t)(kc * 8 + wv) * 32 + lane) << 4));
    b.u4[0] = bp[0];
    b.u4[1] = bp[1];
    a0.u4[0] = *(const v4u*)(meA + ln * 320 + kc * 32 + hf * 8);
    a0.u4[1] = *(const v4u*)(meA + ln * 320 + kc * 32 + 16 + hf * 8);
    a1.u4[0] = *(const v4u*)(meA + (16 + ln) * 320 + kc * 32 + hf * 8);
    a1.u4[1] = *(const v4u*)(meA + (16 + ln) * 320 + kc * 32 + 16 + hf * 8);
    acc0 = __builtin_amdgcn_wmma_f32_16x16x32_bf16(false, a0.bf, false, b.bf,
                                                   (short)0, acc0, false, false);
    acc1 = __builtin_amdgcn_wmma_f32_16x16x32_bf16(false, a1.bf, false, b.bf,
                                                   (short)0, acc1, false, false);
  }
  #pragma unroll
  for (int r = 0; r < 8; ++r) {
    hB[(r + hf * 8) * DD + col]      = f2bf(fmaxf(acc0[r], 0.0f));
    hB[(16 + r + hf * 8) * DD + col] = f2bf(fmaxf(acc1[r], 0.0f));
  }
  __syncthreads();

  // GEMM2: m = h @ W2 + b2, K = 128
  float bv2 = b2[col];
  v8f acc2 = {bv2, bv2, bv2, bv2, bv2, bv2, bv2, bv2};
  v8f acc3 = acc2;
  #pragma unroll
  for (int kc = 0; kc < 4; ++kc) {
    FragAB a0, a1, b;
    const v4u* bp = (const v4u*)(W2p + (((size_t)(kc * 8 + wv) * 32 + lane) << 4));
    b.u4[0] = bp[0];
    b.u4[1] = bp[1];
    a0.u4[0] = *(const v4u*)(hB + ln * DD + kc * 32 + hf * 8);
    a0.u4[1] = *(const v4u*)(hB + ln * DD + kc * 32 + 16 + hf * 8);
    a1.u4[0] = *(const v4u*)(hB + (16 + ln) * DD + kc * 32 + hf * 8);
    a1.u4[1] = *(const v4u*)(hB + (16 + ln) * DD + kc * 32 + 16 + hf * 8);
    acc2 = __builtin_amdgcn_wmma_f32_16x16x32_bf16(false, a0.bf, false, b.bf,
                                                   (short)0, acc2, false, false);
    acc3 = __builtin_amdgcn_wmma_f32_16x16x32_bf16(false, a1.bf, false, b.bf,
                                                   (short)0, acc3, false, false);
  }
  // fused segment-sum into Msum[dst]
  #pragma unroll
  for (int r = 0; r < 8; ++r) {
    int row0 = r + hf * 8;
    int row1 = 16 + row0;
    if (e0 + row0 < NEDG)
      atomicAdd(Msum + (size_t)dIdx[row0] * DD + col, acc2[r]);
    if (e0 + row1 < NEDG)
      atomicAdd(Msum + (size_t)dIdx[row1] * DD + col, acc3[r]);
  }
}

// ---------------- node update: x = relu(LN(relu([x|M]@Wu+bu) + x)) ----------------
__global__ void update_k(float* __restrict__ x, const float* __restrict__ Msum,
                         const float* __restrict__ cntd,
                         const float* __restrict__ wu, const float* __restrict__ bu,
                         const float* __restrict__ lg, const float* __restrict__ lb) {
  __shared__ float xv[DD], Mv[DD], redA[DD], redB[DD];
  int s = blockIdx.x, t = threadIdx.x;
  float xin = x[(size_t)s * DD + t];
  xv[t] = xin;
  Mv[t] = Msum[(size_t)s * DD + t] / fmaxf(cntd[s], 1.0f);
  __syncthreads();
  float acc = bu[t];
  for (int k = 0; k < DD; ++k) acc += xv[k] * wu[k * DD + t];
  for (int k = 0; k < DD; ++k) acc += Mv[k] * wu[(DD + k) * DD + t];
  acc = fmaxf(acc, 0.0f) + xin;  // residual (z == x at iteration start)
  float2 r = block_sum2_128(acc, acc * acc, redA, redB, t);
  float m = r.x * (1.0f / 128.0f);
  float v = fmaxf(r.y * (1.0f / 128.0f) - m * m, 0.0f);
  float y = (acc - m) * rsqrtf(v + LEPS) * lg[t] + lb[t];
  x[(size_t)s * DD + t] = fmaxf(y, 0.0f);
}

// ---------------- classifier + emit (logits | z) ----------------
__global__ void final_k(const float* __restrict__ x, const float* __restrict__ cw,
                        const float* __restrict__ cb, float* __restrict__ out) {
  __shared__ float xv[DD];
  int s = blockIdx.x, t = threadIdx.x;
  float v = x[(size_t)s * DD + t];
  xv[t] = v;
  out[(size_t)NSUP * 13 + (size_t)s * DD + t] = v;  // z
  __syncthreads();
  if (t < 13) {
    float acc = cb[t];
    for (int k = 0; k < DD; ++k) acc += xv[k] * cw[k * 13 + t];
    out[(size_t)s * 13 + t] = acc;
  }
}

extern "C" void kernel_launch(void* const* d_in, const int* in_sizes, int n_in,
                              void* d_out, int out_size, void* d_ws, size_t ws_size,
                              hipStream_t stream) {
  (void)in_sizes; (void)n_in; (void)out_size; (void)ws_size;
  const float* pos    = (const float*)d_in[0];
  const float* pfeat  = (const float*)d_in[1];
  const int*   sidx   = (const int*)d_in[2];
  const int*   esrc   = (const int*)d_in[3];
  const int*   edst   = esrc + NEDG;
  const float* enc_w1 = (const float*)d_in[4];
  const float* enc_b1 = (const float*)d_in[5];
  const float* enc_g1 = (const float*)d_in[6];
  const float* enc_be1= (const float*)d_in[7];
  const float* enc_w2 = (const float*)d_in[8];
  const float* enc_b2 = (const float*)d_in[9];
  const float* enc_g2 = (const float*)d_in[10];
  const float* enc_be2= (const float*)d_in[11];
  const float* pe_w1  = (const float*)d_in[12];
  const float* pe_b1  = (const float*)d_in[13];
  const float* pe_w2  = (const float*)d_in[14];
  const float* pe_b2  = (const float*)d_in[15];
  const float* pm_w1  = (const float*)d_in[16];
  const float* pm_b1  = (const float*)d_in[17];
  const float* pm_w2  = (const float*)d_in[18];
  const float* pm_b2  = (const float*)d_in[19];
  const float* pu_w1  = (const float*)d_in[20];
  const float* pu_b1  = (const float*)d_in[21];
  const float* ln_g   = (const float*)d_in[22];
  const float* ln_b   = (const float*)d_in[23];
  const float* cls_w  = (const float*)d_in[24];
  const float* cls_b  = (const float*)d_in[25];
  float* out = (float*)d_out;

  char* p = (char*)d_ws;
  auto carve = [&](size_t bytes) -> void* {
    void* r = (void*)p;
    p += (bytes + 255) & ~(size_t)255;
    return r;
  };
  float* csum  = (float*)carve((size_t)NSUP * 4 * sizeof(float));
  float* cent  = (float*)carve((size_t)NSUP * 3 * sizeof(float));
  float* cntd  = (float*)carve((size_t)NSUP * sizeof(float));
  float* x     = (float*)carve((size_t)NSUP * DD * sizeof(float));
  unsigned short* xbf = (unsigned short*)carve((size_t)NSUP * DD * 2);
  float* Msum  = (float*)carve((size_t)NSUP * DD * sizeof(float));
  unsigned short* ebf = (unsigned short*)carve((size_t)NEDG * DEE * 2);
  unsigned short* W1p = (unsigned short*)carve((size_t)3 * 320 * DD * 2);
  unsigned short* W2p = (unsigned short*)carve((size_t)3 * DD * DD * 2);

  hipMemsetAsync(csum, 0, (size_t)NSUP * 4 * sizeof(float), stream);
  hipMemsetAsync(cntd, 0, (size_t)NSUP * sizeof(float), stream);
  hipMemsetAsync(x,    0, (size_t)NSUP * DD * sizeof(float), stream);

  cent_acc_k<<<(NPTS + 255) / 256, 256, 0, stream>>>(pos, sidx, csum);
  cent_fin_k<<<(NSUP + 255) / 256, 256, 0, stream>>>(csum, cent);
  deg_acc_k<<<(NEDG + 255) / 256, 256, 0, stream>>>(edst, cntd);

  encoder_k<<<NPTS, 128, 0, stream>>>(pos, pfeat, sidx, cent,
                                      enc_w1, enc_b1, enc_g1, enc_be1,
                                      enc_w2, enc_b2, enc_g2, enc_be2, x);
  tobf_k<<<(NSUP * DD + 255) / 256, 256, 0, stream>>>(x, xbf, NSUP * DD);

  for (int l = 0; l < 3; ++l) {
    packw_k<<<(320 * DD + 255) / 256, 256, 0, stream>>>(pm_w1 + (size_t)l * 320 * DD,
                                                        W1p + (size_t)l * 320 * DD, 320);
    packw_k<<<(DD * DD + 255) / 256, 256, 0, stream>>>(pm_w2 + (size_t)l * DD * DD,
                                                       W2p + (size_t)l * DD * DD, DD);
  }

  for (int l = 0; l < 3; ++l) {
    edgefeat_k<<<(NEDG + 3) / 4, 256, 0, stream>>>(esrc, edst, cent,
                                                   pe_w1 + l * 3 * DEE, pe_b1 + l * DEE,
                                                   pe_w2 + l * DEE * DEE, pe_b2 + l * DEE, ebf);
    hipMemsetAsync(Msum, 0, (size_t)NSUP * DD * sizeof(float), stream);
    edgemsg_k<<<NEDG / 32, 256, 0, stream>>>(esrc, edst, xbf, ebf,
                                             W1p + (size_t)l * 320 * DD, pm_b1 + l * DD,
                                             W2p + (size_t)l * DD * DD, pm_b2 + l * DD, Msum);
    update_k<<<NSUP, 128, 0, stream>>>(x, Msum, cntd,
                                       pu_w1 + (size_t)l * 2 * DD * DD, pu_b1 + l * DD,
                                       ln_g + l * DD, ln_b + l * DD);
    tobf_k<<<(NSUP * DD + 255) / 256, 256, 0, stream>>>(x, xbf, NSUP * DD);
  }

  final_k<<<NSUP, 128, 0, stream>>>(x, cls_w, cls_b, out);
}
